// BysMamba_46042049413921
// MI455X (gfx1250) — matrix operand order repaired and checked by reading, not compiled, not run on hardware
//
#include <hip/hip_runtime.h>
#include <hip/hip_bf16.h>
#include <math.h>

// ---------------------------------------------------------------------------
// Bidirectional Mamba stack for MI455X (gfx1250, wave32, WMMA).
// GEMMs: bf16 in / f32 accumulate, v_wmma_f32_16x16x32_bf16, 32x32 macro-tile
//        per wave (2x2 register blocking, 8 b128 loads -> 4 WMMAs).
// Scan:  3-phase chunked associative scan (32 chunks x 64 steps).
// ---------------------------------------------------------------------------

typedef unsigned short u16;
typedef __attribute__((ext_vector_type(16))) __bf16 v16bf;
typedef __attribute__((ext_vector_type(8)))  float  v8f;

struct __align__(16) Q16 { unsigned x, y, z, w; };
union Frag { v16bf v; Q16 q[2]; };

#define BT 2
#define LT 2048
#define MT (BT * LT)      // 4096 tokens
#define DMODEL 472
#define DPAD 480          // K padded to mult of 32 (bf16)
#define ED2 1888
#define EDM 944
#define EDPAD 960
#define NST 16
#define RK 30
#define DBLN 62
#define DEPTH 8

#define NCHUNK 32
#define CLEN (LT / NCHUNK)            // 64
#define SCN (BT * NCHUNK * EDM)       // 60416 planar stride for scan state

__device__ __forceinline__ u16 f2bf(float f) {
  union { float f; unsigned u; } v; v.f = f;
  unsigned r = v.u + 0x7FFFu + ((v.u >> 16) & 1u);  // round-to-nearest-even
  return (u16)(r >> 16);
}

enum { EPI_PLAIN = 0, EPI_DBL = 1, EPI_SOFTPLUS = 2, EPI_ACCUM = 3 };

// ---------------------------------------------------------------------------
// Epilogue for one 16x16 sub-tile (C/D layout: VGPR r -> M=r / M=r+8).
// ---------------------------------------------------------------------------
template<int EPI>
__device__ __forceinline__ void epi_tile(float* __restrict__ C, u16* __restrict__ obf,
                                         const float* __restrict__ bias, int Nc, int ldc,
                                         int rbase, int col, int half, v8f acc) {
#pragma unroll
  for (int r = 0; r < 8; ++r) {
    const int row = rbase + r + (half << 3);
    const float v = acc[r];
    if (EPI == EPI_PLAIN) {
      if (col < Nc) C[(size_t)row * ldc + col] = v;
    } else if (EPI == EPI_DBL) {
      if (col < Nc) C[(size_t)row * ldc + col] = v;
      if (col < RK)      obf[(size_t)row * 32 + col] = f2bf(v);
      else if (col < 32) obf[(size_t)row * 32 + col] = 0;
    } else if (EPI == EPI_SOFTPLUS) {
      if (col < Nc) {
        float s  = v + bias[col];
        float sp = (s > 20.f) ? s : log1pf(__expf(s));
        C[(size_t)row * ldc + col] = sp;
      }
    } else {  // EPI_ACCUM
      if (col < Nc) C[(size_t)row * ldc + col] += v;
    }
  }
}

// ---------------------------------------------------------------------------
// WMMA GEMM: C[M,N] (+)= A[M,Kpad] @ W[Npad,Kpad]^T.
// W rows are zero-padded to a multiple of 32 -> all loads unconditional
// (no exec-mask churn in the k-loop). 32x32 output macro-tile per wave.
// ---------------------------------------------------------------------------
template<int EPI>
__global__ void wmma_gemm_k(const u16* __restrict__ A, const u16* __restrict__ W,
                            float* __restrict__ C, int Mr, int Nc, int Kpad, int ldc,
                            const float* __restrict__ bias, u16* __restrict__ obf) {
  const int lane   = threadIdx.x & 31;
  const int wave   = threadIdx.x >> 5;
  const int tilesN = (Nc + 31) >> 5;
  const int tiles  = (Mr >> 5) * tilesN;
  const int tile   = blockIdx.x * (blockDim.x >> 5) + wave;
  if (tile >= tiles) return;               // wave-uniform (EXEC all-1 at WMMA)
  const int tm   = (tile / tilesN) << 5;
  const int tn   = (tile % tilesN) << 5;
  const int half = lane >> 4, l15 = lane & 15;

  // A frag: lane<16 row r holds K {0..7,16..23}; lane>=16 holds K {8..15,24..31}
  const u16* ap0 = A + (size_t)(tm + l15) * Kpad + (half << 3);
  const u16* ap1 = ap0 + (size_t)16 * Kpad;
  // B frag: lane n = col n (row n of W); half h holds K {8h..8h+7, 16+8h..16+8h+7}
  const u16* bp0 = W + (size_t)(tn + l15) * Kpad + (half << 3);
  const u16* bp1 = bp0 + (size_t)16 * Kpad;

  v8f acc00 = {0.f,0.f,0.f,0.f,0.f,0.f,0.f,0.f};
  v8f acc01 = acc00, acc10 = acc00, acc11 = acc00;

  for (int k0 = 0; k0 < Kpad; k0 += 32) {
    Frag a0, a1, b0, b1;
    a0.q[0] = *(const Q16*)(ap0 + k0);  a0.q[1] = *(const Q16*)(ap0 + k0 + 16);
    a1.q[0] = *(const Q16*)(ap1 + k0);  a1.q[1] = *(const Q16*)(ap1 + k0 + 16);
    b0.q[0] = *(const Q16*)(bp0 + k0);  b0.q[1] = *(const Q16*)(bp0 + k0 + 16);
    b1.q[0] = *(const Q16*)(bp1 + k0);  b1.q[1] = *(const Q16*)(bp1 + k0 + 16);
    acc00 = __builtin_amdgcn_wmma_f32_16x16x32_bf16(false, a0.v, false, b0.v, (short)0, acc00, false, false);
    acc01 = __builtin_amdgcn_wmma_f32_16x16x32_bf16(false, a0.v, false, b1.v, (short)0, acc01, false, false);
    acc10 = __builtin_amdgcn_wmma_f32_16x16x32_bf16(false, a1.v, false, b0.v, (short)0, acc10, false, false);
    acc11 = __builtin_amdgcn_wmma_f32_16x16x32_bf16(false, a1.v, false, b1.v, (short)0, acc11, false, false);
  }

  epi_tile<EPI>(C, obf, bias, Nc, ldc, tm,      tn + l15,      half, acc00);
  epi_tile<EPI>(C, obf, bias, Nc, ldc, tm,      tn + 16 + l15, half, acc01);
  epi_tile<EPI>(C, obf, bias, Nc, ldc, tm + 16, tn + l15,      half, acc10);
  epi_tile<EPI>(C, obf, bias, Nc, ldc, tm + 16, tn + 16 + l15, half, acc11);
}

// fp32 (rows,K) -> bf16 (rowsPad,Kpad), zero padding both dims
__global__ void cvt_bf16_k(const float* __restrict__ src, u16* __restrict__ dst,
                           int rows, int rowsPad, int K, int Kpad) {
  long long t = (long long)blockIdx.x * blockDim.x + threadIdx.x;
  if (t >= (long long)rowsPad * Kpad) return;
  int r = (int)(t / Kpad), c = (int)(t % Kpad);
  dst[t] = (r < rows && c < K) ? f2bf(src[(size_t)r * K + c]) : (u16)0;
}

// patch embed
__global__ void patch_k(const float* __restrict__ x, const float* __restrict__ pw,
                        const float* __restrict__ pb, float* __restrict__ h) {
  int t = blockIdx.x * blockDim.x + threadIdx.x;
  if (t >= MT * DMODEL) return;
  int v = t % DMODEL, token = t / DMODEL;
  float acc = pb[v];
#pragma unroll
  for (int j = 0; j < 9; ++j) acc += x[(size_t)token * 9 + j] * pw[(size_t)v * 9 + j];
  h[t] = acc;
}

// zero K-pad columns of y_bf (scan only writes e<944)
__global__ void pad_ybf_k(u16* __restrict__ y_bf) {
  int t = blockIdx.x * blockDim.x + threadIdx.x;
  if (t >= MT * 16) return;
  y_bf[(size_t)(t >> 4) * EDPAD + EDM + (t & 15)] = 0;
}

// depthwise causal conv (K=4) + bias + SiLU; reverse handles flipped branch
__global__ void conv_silu_k(const float* __restrict__ xz, const float* __restrict__ cw,
                            const float* __restrict__ cb, float* __restrict__ xc,
                            u16* __restrict__ xc_bf, int reverse) {
  int t = blockIdx.x * blockDim.x + threadIdx.x;
  if (t >= MT * EDPAD) return;
  int e = t % EDPAD, token = t / EDPAD;
  if (e >= EDM) { xc_bf[(size_t)token * EDPAD + e] = 0; return; }
  int b = token / LT, l = token % LT;
  float acc = cb[e];
#pragma unroll
  for (int k = 0; k < 4; ++k) {
    int idx = reverse ? (l + 3 - k) : (l - 3 + k);
    if (idx >= 0 && idx < LT)
      acc += cw[(size_t)e * 4 + k] * xz[((size_t)(b * LT + idx)) * ED2 + e];
  }
  float s = acc / (1.f + __expf(-acc));
  xc[(size_t)token * EDM + e]      = s;
  xc_bf[(size_t)token * EDPAD + e] = f2bf(s);
}

// ---------------------------------------------------------------------------
// Chunked associative scan. Processing step s (0..L-1) maps to sequence index
// l = reverse ? L-1-s : s. Chunk c covers s in [c*64, (c+1)*64).
// Planar state layout: X[n*SCN + (b*NCHUNK + c)*EDM + e]  (coalesced in e).
// ---------------------------------------------------------------------------

// Phase 1: per (b,e,chunk): P = prod dA, S = zero-init partial state.
__global__ void scan_p1_k(const float* __restrict__ xc, const float* __restrict__ delta,
                          const float* __restrict__ dbl, const float* __restrict__ Alog,
                          float* __restrict__ Pst, float* __restrict__ Sst, int reverse) {
  int t = blockIdx.x * blockDim.x + threadIdx.x;
  if (t >= SCN) return;
  int e = t % EDM, bc = t / EDM;
  int c = bc % NCHUNK, b = bc / NCHUNK;
  float A[NST], P[NST], S[NST];
#pragma unroll
  for (int n = 0; n < NST; ++n) { A[n] = -__expf(Alog[(size_t)e * NST + n]); P[n] = 1.f; S[n] = 0.f; }
  for (int s0 = 0; s0 < CLEN; ++s0) {
    int s = c * CLEN + s0;
    int l = reverse ? (LT - 1 - s) : s;
    size_t token = (size_t)b * LT + l;
    float d = delta[token * EDM + e], x = xc[token * EDM + e], dx = d * x;
    const float* bcp = dbl + token * DBLN;
#pragma unroll
    for (int n = 0; n < NST; ++n) {
      float dA = __expf(d * A[n]);
      S[n] = dA * S[n] + dx * bcp[RK + n];
      P[n] *= dA;
    }
  }
#pragma unroll
  for (int n = 0; n < NST; ++n) { Pst[(size_t)n * SCN + t] = P[n]; Sst[(size_t)n * SCN + t] = S[n]; }
}

// Phase 2: per (b,e): chain chunks, record chunk-entry states H0.
__global__ void scan_p2_k(const float* __restrict__ Pst, const float* __restrict__ Sst,
                          float* __restrict__ H0) {
  int t = blockIdx.x * blockDim.x + threadIdx.x;
  if (t >= BT * EDM) return;
  int e = t % EDM, b = t / EDM;
  float H[NST];
#pragma unroll
  for (int n = 0; n < NST; ++n) H[n] = 0.f;
  for (int c = 0; c < NCHUNK; ++c) {
    size_t idx = (size_t)(b * NCHUNK + c) * EDM + e;
#pragma unroll
    for (int n = 0; n < NST; ++n) {
      H0[(size_t)n * SCN + idx] = H[n];
      H[n] = Pst[(size_t)n * SCN + idx] * H[n] + Sst[(size_t)n * SCN + idx];
    }
  }
}

// Phase 3: replay each chunk from its true entry state; emit gated bf16 y.
__global__ void scan_p3_k(const float* __restrict__ xz, const float* __restrict__ xc,
                          const float* __restrict__ delta, const float* __restrict__ dbl,
                          const float* __restrict__ Alog, const float* __restrict__ Dp,
                          const float* __restrict__ H0, u16* __restrict__ y_bf, int reverse) {
  int t = blockIdx.x * blockDim.x + threadIdx.x;
  if (t >= SCN) return;
  int e = t % EDM, bc = t / EDM;
  int c = bc % NCHUNK, b = bc / NCHUNK;
  float A[NST], h[NST];
#pragma unroll
  for (int n = 0; n < NST; ++n) {
    A[n] = -__expf(Alog[(size_t)e * NST + n]);
    h[n] = H0[(size_t)n * SCN + t];
  }
  float Dv = Dp[e];
  for (int s0 = 0; s0 < CLEN; ++s0) {
    int s = c * CLEN + s0;
    int l = reverse ? (LT - 1 - s) : s;
    size_t token = (size_t)b * LT + l;
    float d = delta[token * EDM + e], x = xc[token * EDM + e], dx = d * x;
    const float* bcp = dbl + token * DBLN;
    float y = 0.f;
#pragma unroll
    for (int n = 0; n < NST; ++n) {
      float dA = __expf(d * A[n]);
      h[n] = dA * h[n] + dx * bcp[RK + n];
      y += h[n] * bcp[RK + NST + n];
    }
    float z   = xz[token * ED2 + EDM + e];
    float out = (y + Dv * x) * (z / (1.f + __expf(-z)));
    y_bf[token * EDPAD + e] = f2bf(out);
  }
}

// ---------------------------------------------------------------------------
template<int EPI>
static inline void launch_gemm(hipStream_t s, const u16* A, const u16* W, float* C,
                               int Mr, int Nc, int Kpad, int ldc,
                               const float* bias, u16* obf) {
  int tiles  = (Mr >> 5) * ((Nc + 31) >> 5);   // 32x32 macro-tiles
  int blocks = (tiles + 7) / 8;                // 8 waves per block
  wmma_gemm_k<EPI><<<blocks, 256, 0, s>>>(A, W, C, Mr, Nc, Kpad, ldc, bias, obf);
}

extern "C" void kernel_launch(void* const* d_in, const int* in_sizes, int n_in,
                              void* d_out, int out_size, void* d_ws, size_t ws_size,
                              hipStream_t stream) {
  (void)in_sizes; (void)n_in; (void)out_size; (void)ws_size;

  const float* x       = (const float*)d_in[0];
  const float* patch_w = (const float*)d_in[1];
  const float* patch_b = (const float*)d_in[2];
  const float* lm_w    = (const float*)d_in[3];

  struct Pset { const float *inproj, *conv_w, *conv_b, *xproj, *dt_w, *dt_b, *Alog, *D, *outproj; };
  auto pset = [&](int base) {
    Pset p;
    p.inproj  = (const float*)d_in[base + 0];
    p.conv_w  = (const float*)d_in[base + 1];
    p.conv_b  = (const float*)d_in[base + 2];
    p.xproj   = (const float*)d_in[base + 3];
    p.dt_w    = (const float*)d_in[base + 4];
    p.dt_b    = (const float*)d_in[base + 5];
    p.Alog    = (const float*)d_in[base + 6];
    p.D       = (const float*)d_in[base + 7];
    p.outproj = (const float*)d_in[base + 8];
    return p;
  };
  Pset Pin = pset(4), Play = pset(13), Pout = pset(22);

  // ---- workspace bump allocator ----
  char* wsb = (char*)d_ws;
  size_t off = 0;
  auto alloc = [&](size_t bytes) -> void* {
    off = (off + 255) & ~(size_t)255;
    void* p = wsb + off;
    off += bytes;
    return p;
  };
  float* h      = (float*)alloc((size_t)MT * DMODEL * 4);
  u16*   h_bf   = (u16*)  alloc((size_t)MT * DPAD * 2);
  float* xz     = (float*)alloc((size_t)MT * ED2 * 4);
  float* xc     = (float*)alloc((size_t)MT * EDM * 4);
  u16*   xc_bf  = (u16*)  alloc((size_t)MT * EDPAD * 2);
  float* dbl    = (float*)alloc((size_t)MT * DBLN * 4);
  u16*   dbl_bf = (u16*)  alloc((size_t)MT * 32 * 2);
  float* delta  = (float*)alloc((size_t)MT * EDM * 4);
  u16*   y_bf   = (u16*)  alloc((size_t)MT * EDPAD * 2);
  float* scanP  = (float*)alloc((size_t)NST * SCN * 4);
  float* scanS  = (float*)alloc((size_t)NST * SCN * 4);
  float* scanH0 = (float*)alloc((size_t)NST * SCN * 4);

  // bf16 weights; rows padded to mult of 32 so GEMM loads are unconditional
  u16* Wi_in  = (u16*)alloc((size_t)ED2 * DPAD * 2);                // 1888 rows (ok)
  u16* Wi_lay = (u16*)alloc((size_t)DEPTH * ED2 * DPAD * 2);
  u16* Wi_out = (u16*)alloc((size_t)ED2 * DPAD * 2);
  u16* Wx_in  = (u16*)alloc((size_t)64 * EDPAD * 2);                // 62 -> 64 rows
  u16* Wx_lay = (u16*)alloc((size_t)DEPTH * 64 * EDPAD * 2);
  u16* Wx_out = (u16*)alloc((size_t)64 * EDPAD * 2);
  u16* Wd_in  = (u16*)alloc((size_t)960 * 32 * 2);                  // 944 -> 960 rows
  u16* Wd_lay = (u16*)alloc((size_t)DEPTH * 960 * 32 * 2);
  u16* Wd_out = (u16*)alloc((size_t)960 * 32 * 2);
  u16* Wo_in  = (u16*)alloc((size_t)480 * EDPAD * 2);               // 472 -> 480 rows
  u16* Wo_lay = (u16*)alloc((size_t)DEPTH * 480 * EDPAD * 2);
  u16* Wo_out = (u16*)alloc((size_t)480 * EDPAD * 2);
  u16* Wlm    = (u16*)alloc((size_t)480 * DPAD * 2);

  auto cvt = [&](const float* src, u16* dst, int rows, int rowsPad, int K, int Kpad) {
    long long tot = (long long)rowsPad * Kpad;
    int blocks = (int)((tot + 255) / 256);
    cvt_bf16_k<<<blocks, 256, 0, stream>>>(src, dst, rows, rowsPad, K, Kpad);
  };

  // ---- weight conversion (row+K zero padding) ----
  cvt(Pin.inproj,  Wi_in,  ED2,         ED2,         DMODEL, DPAD);
  cvt(Play.inproj, Wi_lay, DEPTH * ED2, DEPTH * ED2, DMODEL, DPAD);  // 1888 | 32
  cvt(Pout.inproj, Wi_out, ED2,         ED2,         DMODEL, DPAD);
  cvt(Pin.xproj,   Wx_in,  DBLN, 64,  EDM, EDPAD);
  cvt(Pout.xproj,  Wx_out, DBLN, 64,  EDM, EDPAD);
  cvt(Pin.dt_w,    Wd_in,  EDM,  960, RK,  32);
  cvt(Pout.dt_w,   Wd_out, EDM,  960, RK,  32);
  cvt(Pin.outproj, Wo_in,  DMODEL, 480, EDM, EDPAD);
  cvt(Pout.outproj,Wo_out, DMODEL, 480, EDM, EDPAD);
  cvt(lm_w,        Wlm,    DMODEL, 480, DMODEL, DPAD);
  for (int i = 0; i < DEPTH; ++i) {  // per-layer (row padding breaks stacked cvt)
    cvt(Play.xproj   + (size_t)i * DBLN * EDM,   Wx_lay + (size_t)i * 64 * EDPAD,  DBLN,   64,  EDM,    EDPAD);
    cvt(Play.dt_w    + (size_t)i * EDM * RK,     Wd_lay + (size_t)i * 960 * 32,    EDM,    960, RK,     32);
    cvt(Play.outproj + (size_t)i * DMODEL * EDM, Wo_lay + (size_t)i * 480 * EDPAD, DMODEL, 480, EDM,    EDPAD);
  }

  // ---- patch embed + y_bf K-pad init ----
  patch_k<<<(MT * DMODEL + 255) / 256, 256, 0, stream>>>(x, patch_w, patch_b, h);
  pad_ybf_k<<<(MT * 16 + 255) / 256, 256, 0, stream>>>(y_bf);

  auto cvt_h = [&]() { cvt(h, h_bf, MT, MT, DMODEL, DPAD); };

  auto run_block = [&](const float* cw, const float* cb, const float* Alog, const float* Dp,
                       const float* dtb, const u16* Wi, const u16* Wx, const u16* Wd,
                       const u16* Wo, int rev) {
    launch_gemm<EPI_PLAIN>(stream, h_bf, Wi, xz, MT, ED2, DPAD, ED2, nullptr, nullptr);
    conv_silu_k<<<(MT * EDPAD + 255) / 256, 256, 0, stream>>>(xz, cw, cb, xc, xc_bf, rev);
    launch_gemm<EPI_DBL>(stream, xc_bf, Wx, dbl, MT, DBLN, EDPAD, DBLN, nullptr, dbl_bf);
    launch_gemm<EPI_SOFTPLUS>(stream, dbl_bf, Wd, delta, MT, EDM, 32, EDM, dtb, nullptr);
    scan_p1_k<<<(SCN + 255) / 256, 256, 0, stream>>>(xc, delta, dbl, Alog, scanP, scanS, rev);
    scan_p2_k<<<(BT * EDM + 127) / 128, 128, 0, stream>>>(scanP, scanS, scanH0);
    scan_p3_k<<<(SCN + 255) / 256, 256, 0, stream>>>(xz, xc, delta, dbl, Alog, Dp, scanH0,
                                                     y_bf, rev);
    launch_gemm<EPI_ACCUM>(stream, y_bf, Wo, h, MT, DMODEL, EDPAD, DMODEL, nullptr, nullptr);
  };

  // h = h + bidir(h, p_in): both directions read the same h_bf snapshot.
  cvt_h();
  run_block(Pin.conv_w, Pin.conv_b, Pin.Alog, Pin.D, Pin.dt_b, Wi_in, Wx_in, Wd_in, Wo_in, 0);
  run_block(Pin.conv_w, Pin.conv_b, Pin.Alog, Pin.D, Pin.dt_b, Wi_in, Wx_in, Wd_in, Wo_in, 1);

  // 8 residual layers
  for (int i = 0; i < DEPTH; ++i) {
    cvt_h();
    run_block(Play.conv_w + (size_t)i * EDM * 4, Play.conv_b + (size_t)i * EDM,
              Play.Alog + (size_t)i * EDM * NST, Play.D + (size_t)i * EDM,
              Play.dt_b + (size_t)i * EDM,
              Wi_lay + (size_t)i * ED2 * DPAD, Wx_lay + (size_t)i * 64 * EDPAD,
              Wd_lay + (size_t)i * 960 * 32,   Wo_lay + (size_t)i * 480 * EDPAD, 0);
  }

  // h = h + bidir(h, p_out)
  cvt_h();
  run_block(Pout.conv_w, Pout.conv_b, Pout.Alog, Pout.D, Pout.dt_b, Wi_out, Wx_out, Wd_out, Wo_out, 0);
  run_block(Pout.conv_w, Pout.conv_b, Pout.Alog, Pout.D, Pout.dt_b, Wi_out, Wx_out, Wd_out, Wo_out, 1);

  // lm_head -> d_out (fp32)
  cvt_h();
  launch_gemm<EPI_PLAIN>(stream, h_bf, Wlm, (float*)d_out, MT, DMODEL, DPAD, DMODEL,
                         nullptr, nullptr);
}
